// ConvexSHLossColBERT_74466142978309
// MI455X (gfx1250) — matrix operand order repaired and checked by reading, not compiled, not run on hardware
//
#include <hip/hip_runtime.h>
#include <math.h>

// ConvexSHLossColBERT — fused masked-l2norm + bf16 WMMA maxsim + focal loss.
// B=128, LQ=32, LD=256, D=128, NWAY=8. Bandwidth-bound: 128 MiB doc read once.

constexpr int BATCH = 128;
constexpr int LQ    = 32;
constexpr int LDOC  = 256;
constexpr int DIM   = 128;
constexpr int NWAY  = 8;
constexpr float ALPHA_C = 0.2f;
constexpr float GAMMA_C = 2.0f;

// LDS strides (in bf16 elements). 136 = 128 + 8 pad: row byte-stride 272 = 68
// dwords -> bank(row) = 4*row mod 64, conflict-free across a 16-lane group,
// and every fragment address stays 16-byte aligned (136 % 8 == 0).
constexpr int QS = 136;
constexpr int DS = 136;

typedef __attribute__((ext_vector_type(16))) __bf16 v16bf;
typedef __attribute__((ext_vector_type(8)))  __bf16 v8bf;
typedef __attribute__((ext_vector_type(8)))  float  v8f;

union V16U { v16bf v; v8bf h[2]; };

__global__ __launch_bounds__(256) void colbert_score_kernel(
    const float* __restrict__ qreps,   // [B][LQ][D]
    const float* __restrict__ dreps,   // [NWAY][B][LD][D]
    const float* __restrict__ dmasks,  // [NWAY][B][LD]
    float* __restrict__ scores)        // [B][NWAY]
{
  const int b    = blockIdx.x / NWAY;
  const int n    = blockIdx.x % NWAY;
  const int t    = threadIdx.x;
  const int lane = t & 31;
  const int w    = t >> 5;        // wave id 0..7
  const int half = (lane >> 4) & 1;
  const int l16  = lane & 15;

  __shared__ __align__(16) __bf16 sQ[LQ * QS];        // ~8.5 KB
  __shared__ __align__(16) __bf16 sD[128 * DS];       // ~34 KB (one 128-row chunk)
  __shared__ float sRed[8][32];                       // per-wave row maxes

  // ---- Phase 1: l2-normalize query tile (32 x 128) -> LDS bf16 ----
  {
    const int row = t >> 3;       // 8 threads per row
    const int t8  = t & 7;
    const float* src = qreps + ((size_t)b * LQ + row) * DIM + t8 * 16;
    float vv[16];
#pragma unroll
    for (int i = 0; i < 4; ++i) {
      float4 f = ((const float4*)src)[i];
      vv[4*i+0] = f.x; vv[4*i+1] = f.y; vv[4*i+2] = f.z; vv[4*i+3] = f.w;
    }
    float ss = 0.f;
#pragma unroll
    for (int i = 0; i < 16; ++i) ss += vv[i] * vv[i];
    ss += __shfl_xor(ss, 1, 32);
    ss += __shfl_xor(ss, 2, 32);
    ss += __shfl_xor(ss, 4, 32);
    const float sc = 1.0f / fmaxf(sqrtf(ss), 1e-12f);
    __bf16* dst = &sQ[row * QS + t8 * 16];
#pragma unroll
    for (int i = 0; i < 16; ++i) dst[i] = (__bf16)(vv[i] * sc);
  }

  float rm[2][8];  // running row-max per lane: [mtile][slot]
#pragma unroll
  for (int mt = 0; mt < 2; ++mt)
#pragma unroll
    for (int r = 0; r < 8; ++r) rm[mt][r] = -3.0e38f;

  for (int chunk = 0; chunk < 2; ++chunk) {
    // ---- Phase 2: mask * l2norm doc chunk (128 x 128) -> LDS bf16 ----
#pragma unroll
    for (int rp = 0; rp < 4; ++rp) {
      const int rl = rp * 32 + (t >> 3);          // row within chunk
      const int rg = chunk * 128 + rl;            // row within doc
      const int t8 = t & 7;
      const size_t rowbase = ((size_t)n * BATCH + b) * LDOC + rg;
      const float m = dmasks[rowbase];
      const float* src = dreps + rowbase * DIM + t8 * 16;
      float vv[16];
#pragma unroll
      for (int i = 0; i < 4; ++i) {
        float4 f = ((const float4*)src)[i];
        vv[4*i+0] = f.x * m; vv[4*i+1] = f.y * m;
        vv[4*i+2] = f.z * m; vv[4*i+3] = f.w * m;
      }
      float ss = 0.f;
#pragma unroll
      for (int i = 0; i < 16; ++i) ss += vv[i] * vv[i];
      ss += __shfl_xor(ss, 1, 32);
      ss += __shfl_xor(ss, 2, 32);
      ss += __shfl_xor(ss, 4, 32);
      const float sc = 1.0f / fmaxf(sqrtf(ss), 1e-12f);  // masked row -> zeros
      __bf16* dst = &sD[rl * DS + t8 * 16];
#pragma unroll
      for (int i = 0; i < 16; ++i) dst[i] = (__bf16)(vv[i] * sc);
    }
    __syncthreads();

    // ---- Phase 3: WMMA GEMM. Wave w owns N-tile (doc cols w*16..w*16+15). ----
    v8f acc0 = {};
    v8f acc1 = {};
#pragma unroll
    for (int kk = 0; kk < 4; ++kk) {
      // B fragment (32x16, K=dim, N=doc pos): lane<16 -> K {0..7, 16..23} of col l16,
      // lane>=16 -> K {8..15, 24..31}.
      const int bq = (w * 16 + l16) * DS + kk * 32 + half * 8;
      V16U bf;
      bf.h[0] = *(const v8bf*)&sD[bq];
      bf.h[1] = *(const v8bf*)&sD[bq + 16];
      // A fragments (16x32): lane<16 -> row M=l16, K {0..7, 16..23};
      // lane>=16 -> same row, K {8..15, 24..31}.
      const int a0 = (l16) * QS + kk * 32 + half * 8;
      V16U af0;
      af0.h[0] = *(const v8bf*)&sQ[a0];
      af0.h[1] = *(const v8bf*)&sQ[a0 + 16];
      const int a1 = (16 + l16) * QS + kk * 32 + half * 8;
      V16U af1;
      af1.h[0] = *(const v8bf*)&sQ[a1];
      af1.h[1] = *(const v8bf*)&sQ[a1 + 16];

      acc0 = __builtin_amdgcn_wmma_f32_16x16x32_bf16(
          false, af0.v, false, bf.v, (short)0, acc0, false, false);
      acc1 = __builtin_amdgcn_wmma_f32_16x16x32_bf16(
          false, af1.v, false, bf.v, (short)0, acc1, false, false);
    }
#pragma unroll
    for (int r = 0; r < 8; ++r) {
      rm[0][r] = fmaxf(rm[0][r], acc0[r]);
      rm[1][r] = fmaxf(rm[1][r], acc1[r]);
    }
    __syncthreads();  // before chunk 1 overwrites sD
  }

  // ---- Phase 4: max over the wave's 16 N columns (16-lane butterflies). ----
#pragma unroll
  for (int mt = 0; mt < 2; ++mt) {
#pragma unroll
    for (int r = 0; r < 8; ++r) {
      float v = rm[mt][r];
      v = fmaxf(v, __shfl_xor(v, 1, 32));
      v = fmaxf(v, __shfl_xor(v, 2, 32));
      v = fmaxf(v, __shfl_xor(v, 4, 32));
      v = fmaxf(v, __shfl_xor(v, 8, 32));
      rm[mt][r] = v;
    }
  }
  // C/D layout: slot r of lanes 0-15 is M = mt*16 + r; lanes 16-31 is M = mt*16+8+r
  if (l16 == 0) {
#pragma unroll
    for (int r = 0; r < 8; ++r) {
      sRed[w][half * 8 + r]      = rm[0][r];
      sRed[w][16 + half * 8 + r] = rm[1][r];
    }
  }
  __syncthreads();

  // ---- Phase 5: cross-wave max per query row, then sum over the 32 rows. ----
  if (t < 32) {
    float v = sRed[0][t];
#pragma unroll
    for (int ww = 1; ww < 8; ++ww) v = fmaxf(v, sRed[ww][t]);
    v += __shfl_xor(v, 1, 32);
    v += __shfl_xor(v, 2, 32);
    v += __shfl_xor(v, 4, 32);
    v += __shfl_xor(v, 8, 32);
    v += __shfl_xor(v, 16, 32);
    if (t == 0) scores[blockIdx.x] = v;
  }
}

__global__ __launch_bounds__(128) void colbert_loss_kernel(
    const float* __restrict__ scores,  // [B][NWAY]
    const float* __restrict__ labels,  // [B][3*NWAY]
    float* __restrict__ out)           // [1]
{
  const int b = threadIdx.x;           // 128 threads, one per batch row
  float s[NWAY], teacher[NWAY], ranks[NWAY], wm[NWAY];
  const float* lr = labels + (size_t)b * 3 * NWAY;
#pragma unroll
  for (int n = 0; n < NWAY; ++n) {
    s[n]       = scores[b * NWAY + n];
    teacher[n] = lr[n];
    ranks[n]   = lr[NWAY + n];
    wm[n]      = lr[2 * NWAY + n];
  }
  float mx = s[0];
#pragma unroll
  for (int n = 1; n < NWAY; ++n) mx = fmaxf(mx, s[n]);
  float e[NWAY], es = 0.f;
#pragma unroll
  for (int n = 0; n < NWAY; ++n) { e[n] = expf(s[n] - mx); es += e[n]; }
  const float inv_es = 1.0f / es;
  const float r0inv  = 1.0f / ranks[0];

  float lsum = 0.f;
#pragma unroll
  for (int n = 0; n < NWAY; ++n) {
    const float p    = e[n] * inv_es;
    const float p2   = p * wm[n] + (1.0f - p) * (1.0f - wm[n]);
    const float ti   = teacher[n] * wm[n] + (1.0f - teacher[n]) * (1.0f - wm[n]);
    const float loss = teacher[n] * (logf(ti) - logf(p2));
    const float wgt  = GAMMA_C - ALPHA_C * (1.0f / ranks[n] - r0inv);
    lsum += loss * wm[n] * powf(1.0f - p2, wgt)
          + loss * (1.0f - wm[n]) * powf(p2, wgt);
  }

  // block reduction: 4 waves of 32
  lsum += __shfl_xor(lsum, 1, 32);
  lsum += __shfl_xor(lsum, 2, 32);
  lsum += __shfl_xor(lsum, 4, 32);
  lsum += __shfl_xor(lsum, 8, 32);
  lsum += __shfl_xor(lsum, 16, 32);
  __shared__ float part[4];
  if ((threadIdx.x & 31) == 0) part[threadIdx.x >> 5] = lsum;
  __syncthreads();
  if (threadIdx.x == 0)
    out[0] = (part[0] + part[1] + part[2] + part[3]) / (float)(BATCH * NWAY);
}

extern "C" void kernel_launch(void* const* d_in, const int* in_sizes, int n_in,
                              void* d_out, int out_size, void* d_ws, size_t ws_size,
                              hipStream_t stream) {
  const float* qreps  = (const float*)d_in[0];
  const float* dreps  = (const float*)d_in[1];
  const float* dmasks = (const float*)d_in[2];
  const float* labels = (const float*)d_in[3];
  float* scores = (float*)d_ws;  // 128*8 floats of scratch

  colbert_score_kernel<<<BATCH * NWAY, 256, 0, stream>>>(qreps, dreps, dmasks, scores);
  colbert_loss_kernel<<<1, BATCH, 0, stream>>>(scores, labels, (float*)d_out);
}
